// LocalCausalAttention_22488448762150
// MI455X (gfx1250) — matrix operand order repaired and checked by reading, not compiled, not run on hardware
//
#include <hip/hip_runtime.h>
#include <math.h>

typedef _Float16 f16;
typedef __attribute__((ext_vector_type(16))) _Float16 v16h;
typedef __attribute__((ext_vector_type(8)))  _Float16 v8h;
typedef __attribute__((ext_vector_type(4)))  _Float16 v4h;
typedef __attribute__((ext_vector_type(8)))  float    v8f;

// Problem constants (match reference setup_inputs)
#define BB   4
#define TT   4096
#define HH   16
#define DD   64
#define WW   64
#define NB   (TT / WW)

// LDS padded strides (halves)
#define QS_STRIDE 72    // 64 + 8
#define KS_STRIDE 72
#define VT_STRIDE 136   // 128 + 8
#define PS_STRIDE 136

static __device__ __forceinline__ v16h join16(v8h lo, v8h hi) {
    return __builtin_shufflevector(lo, hi, 0,1,2,3,4,5,6,7,8,9,10,11,12,13,14,15);
}

__global__ __launch_bounds__(128) void
lca_wmma_kernel(const float* __restrict__ q, const float* __restrict__ k,
                const float* __restrict__ v, const int* __restrict__ amask,
                float* __restrict__ out)
{
    __shared__ f16  Qs[64 * QS_STRIDE];    //  9216 B
    __shared__ f16  Ks[128 * KS_STRIDE];   // 18432 B
    __shared__ f16  VsT[64 * VT_STRIDE];   // 17408 B  (transposed: [d][key])
    __shared__ f16  Ps[64 * PS_STRIDE];    // 17408 B
    __shared__ float qmS[64];
    __shared__ int   kmS[128];

    const int j   = blockIdx.x;   // block index along t
    const int hh  = blockIdx.y;
    const int bb  = blockIdx.z;
    const int tid = threadIdx.x;
    const int lane = tid & 31;
    const int wv   = tid >> 5;          // wave id 0..3 -> M-tile
    const int t0   = j * WW;

    const long rowStride = (long)HH * DD;   // floats between consecutive t
    const float* qb = q + ((long)bb * TT + t0) * rowStride + (long)hh * DD;
    const float* kb = k + ((long)bb * TT) * rowStride + (long)hh * DD;
    const float* vb = v + ((long)bb * TT) * rowStride + (long)hh * DD;

    // ---- masks ----
    {
        int tcur = t0 - WW + tid;                       // key-window position tid
        kmS[tid] = (tcur >= 0) ? amask[(long)bb * TT + tcur] : 0;
        if (tid < 64) qmS[tid] = (float)amask[(long)bb * TT + t0 + tid];
    }

    // ---- load Q (64x64) as f16, row-major ----
    for (int e = tid; e < 64 * 16; e += 128) {
        int r = e >> 4, c4 = e & 15;
        float4 val = *((const float4*)(qb + (long)r * rowStride) + c4);
        v4h hq = { (f16)val.x, (f16)val.y, (f16)val.z, (f16)val.w };
        *(v4h*)&Qs[r * QS_STRIDE + c4 * 4] = hq;
    }
    // ---- load K window (128x64) row-major, V window transposed [64][128] ----
    for (int e = tid; e < 128 * 16; e += 128) {
        int m = e >> 4, c4 = e & 15;
        int tcur = t0 - WW + m;
        float4 kv = {0.f, 0.f, 0.f, 0.f}, vv = {0.f, 0.f, 0.f, 0.f};
        if (tcur >= 0) {
            kv = *((const float4*)(kb + (long)tcur * rowStride) + c4);
            vv = *((const float4*)(vb + (long)tcur * rowStride) + c4);
        }
        v4h hk = { (f16)kv.x, (f16)kv.y, (f16)kv.z, (f16)kv.w };
        *(v4h*)&Ks[m * KS_STRIDE + c4 * 4] = hk;
        int dd0 = c4 * 4;
        VsT[(dd0 + 0) * VT_STRIDE + m] = (f16)vv.x;
        VsT[(dd0 + 1) * VT_STRIDE + m] = (f16)vv.y;
        VsT[(dd0 + 2) * VT_STRIDE + m] = (f16)vv.z;
        VsT[(dd0 + 3) * VT_STRIDE + m] = (f16)vv.w;
    }
    __syncthreads();

    const int col   = lane & 15;          // N column / A row within tile
    const int hsel  = lane >> 4;          // 0: lanes 0-15, 1: lanes 16-31
    const int aoff  = hsel ? 8 : 0;       // A-fragment K sub-offset
    const int khalf = hsel ? 8 : 0;       // B-fragment K sub-offset

    // ================= S = Q * K^T  (scores [16 x 128] per wave) ==========
    v8f acc[8] = {};
    for (int kc = 0; kc < 2; ++kc) {
        const f16* qrow = &Qs[(16 * wv + col) * QS_STRIDE];
        int o = kc * 32 + aoff;
        v16h afr = join16(*(const v8h*)&qrow[o], *(const v8h*)&qrow[o + 16]);
        for (int n = 0; n < 8; ++n) {
            const f16* krow = &Ks[(16 * n + col) * KS_STRIDE];
            int ob = kc * 32 + khalf;
            v16h bfr = join16(*(const v8h*)&krow[ob], *(const v8h*)&krow[ob + 16]);
            acc[n] = __builtin_amdgcn_wmma_f32_16x16x32_f16(
                false, afr, false, bfr, (short)0, acc[n], false, false);
        }
    }

    // ================= masked softmax (register, 16-lane reductions) ======
    const float scale = 0.125f;   // 1/sqrt(64)
    for (int r = 0; r < 8; ++r) {
        const int rl = 16 * wv + r + 8 * hsel;      // local query row 0..63
        float rmax = -INFINITY;
        for (int n = 0; n < 8; ++n) {
            int m = 16 * n + col;
            bool valid = (m > rl) && (m <= rl + WW) && (kmS[m] != 0);
            float s = valid ? acc[n][r] * scale : -INFINITY;
            acc[n][r] = s;
            rmax = fmaxf(rmax, s);
        }
        for (int dlt = 1; dlt < 16; dlt <<= 1)
            rmax = fmaxf(rmax, __shfl_xor(rmax, dlt, 32));

        float rsum = 0.f;
        if (rmax > -INFINITY) {
            for (int n = 0; n < 8; ++n) {
                float p = __expf(acc[n][r] - rmax);   // invalid (-inf) -> 0
                acc[n][r] = p;
                rsum += p;
            }
        } else {
            for (int n = 0; n < 8; ++n) acc[n][r] = 0.f;
        }
        for (int dlt = 1; dlt < 16; dlt <<= 1)
            rsum += __shfl_xor(rsum, dlt, 32);

        float rinv = (rsum > 0.f) ? (qmS[rl] / rsum) : 0.f;
        for (int n = 0; n < 8; ++n)
            Ps[rl * PS_STRIDE + 16 * n + col] = (f16)(acc[n][r] * rinv);
    }
    __syncthreads();

    // ================= O = P * V  ([16 x 64] per wave) ====================
    v8f oacc[4] = {};
    for (int kc = 0; kc < 4; ++kc) {
        const f16* prow = &Ps[(16 * wv + col) * PS_STRIDE];
        int o = kc * 32 + aoff;
        v16h afr = join16(*(const v8h*)&prow[o], *(const v8h*)&prow[o + 16]);
        for (int n = 0; n < 4; ++n) {
            const f16* vrow = &VsT[(16 * n + col) * VT_STRIDE];
            int ob = kc * 32 + khalf;
            v16h bfr = join16(*(const v8h*)&vrow[ob], *(const v8h*)&vrow[ob + 16]);
            oacc[n] = __builtin_amdgcn_wmma_f32_16x16x32_f16(
                false, afr, false, bfr, (short)0, oacc[n], false, false);
        }
    }

    // ---- store O per C/D layout: lane holds column (16n+col), rows r(+8*hsel)
    float* ob = out + ((long)bb * TT + t0) * rowStride + (long)hh * DD;
    for (int n = 0; n < 4; ++n) {
        for (int r = 0; r < 8; ++r) {
            int rl = 16 * wv + r + 8 * hsel;
            ob[(long)rl * rowStride + 16 * n + col] = oacc[n][r];
        }
    }
}

extern "C" void kernel_launch(void* const* d_in, const int* in_sizes, int n_in,
                              void* d_out, int out_size, void* d_ws, size_t ws_size,
                              hipStream_t stream) {
    const float* q = (const float*)d_in[0];
    const float* k = (const float*)d_in[1];
    const float* v = (const float*)d_in[2];
    const int*   m = (const int*)d_in[3];
    float* out = (float*)d_out;
    (void)in_sizes; (void)n_in; (void)out_size; (void)d_ws; (void)ws_size;

    dim3 grid(NB, HH, BB);   // (64, 16, 4) workgroups
    dim3 block(128);         // 4 waves of 32
    lca_wmma_kernel<<<grid, block, 0, stream>>>(q, k, v, m, out);
}
